// MultiHeadAttention_88192858456345
// MI455X (gfx1250) — matrix operand order repaired and checked by reading, not compiled, not run on hardware
//
#include <hip/hip_runtime.h>
#include <hip/hip_bf16.h>

// ---------------------------------------------------------------------------
// MHA for MI455X (gfx1250): all GEMMs via v_wmma_f32_16x16x32_bf16 (wave32).
// Pipeline:
//   1) convert weights fp32 -> bf16
//   2) transpose X (B,C,L) -> Xt (B,L,C) bf16
//   3) proj: Qt/Kt (b,l,512) bf16 (Q pre-scaled 1/8), V (b,512,L) bf16
//   4) fused attention: scores (WMMA) -> LDS (16x2048 f32) -> softmax ->
//      write A (536MB, touched once) -> A@V^T (WMMA, v_perm bf16 pack) -> Ht
//   5) out = Wh @ Hcat + bh (WMMA), fp32 to d_out
// ---------------------------------------------------------------------------

#define L_SEQ 2048
#define DM    512
#define NB    4
#define NH    8
#define SP    2052   // LDS score pitch (pad 4 dwords to stagger banks)

typedef __attribute__((ext_vector_type(16))) __bf16 bf16x16;
typedef __attribute__((ext_vector_type(8)))  float  floatx8;

union BF16x16 {
  bf16x16        v;
  unsigned short s[16];
  unsigned int   u[8];
  uint4          q[2];
};

__device__ inline unsigned short f2bfbits(float x) {
  union { float f; unsigned u; } c; c.f = x;
  unsigned u = c.u;
  unsigned r = u + 0x7FFFu + ((u >> 16) & 1u);   // round-to-nearest-even
  return (unsigned short)(r >> 16);
}

// Pack {bf16(f_odd), bf16(f_even)} into one dword with a single v_perm_b32
// (truncation rounding; used for softmax-prob fragments where 2^-8 rel err ok)
__device__ inline unsigned int pk_bf16_trunc(float f_even, float f_odd) {
  return __builtin_amdgcn_perm(__float_as_uint(f_odd), __float_as_uint(f_even),
                               0x07060302u);
}

__device__ inline floatx8 wmma_bf16(const BF16x16& a, const BF16x16& b, floatx8 c) {
  // (neg_a, A, neg_b, B, c_mod, C, reuse_a, reuse_b)
  return __builtin_amdgcn_wmma_f32_16x16x32_bf16(false, a.v, false, b.v,
                                                 (short)0, c, false, false);
}

// --------------------------- weight fp32 -> bf16 ---------------------------
__global__ void wconv_kernel(const float* __restrict__ W,
                             unsigned short* __restrict__ dst, int n) {
  int i = blockIdx.x * blockDim.x + threadIdx.x;
  if (i < n) dst[i] = f2bfbits(W[i]);
}

// ------------------- X (B,C,L) fp32 -> Xt (B,L,C) bf16 ---------------------
__global__ __launch_bounds__(256)
void xpose_kernel(const float* __restrict__ src, unsigned short* __restrict__ dst) {
  __shared__ float tile[32][33];
  int b  = blockIdx.z;
  int l0 = blockIdx.x * 32, c0 = blockIdx.y * 32;
  const float* s = src + (size_t)b * DM * L_SEQ;
  unsigned short* d = dst + (size_t)b * L_SEQ * DM;
  int tx = threadIdx.x & 31;
  int ty = threadIdx.x >> 5;  // 0..7
  #pragma unroll
  for (int j = ty; j < 32; j += 8)
    tile[j][tx] = s[(size_t)(c0 + j) * L_SEQ + l0 + tx];
  __syncthreads();
  #pragma unroll
  for (int j = ty; j < 32; j += 8)
    d[(size_t)(l0 + j) * DM + c0 + tx] = f2bfbits(tile[tx][j]);
}

// ------------------------------- projections -------------------------------
// Computes outT[l,o] = sum_c Xt[l,c] * W[o,c] per 16x16 tile, one wave/tile.
// mode 0: store bf16 transposed dst[b][l][o]  (* scale)      (Qt, Kt)
// mode 1: store bf16 normal     dst[b][o][l]                 (V)
// mode 2: store fp32 normal + bias -> d_out                  (final out)
__global__ __launch_bounds__(256)
void proj_kernel(const unsigned short* __restrict__ Xt,   // (B, L, DM) bf16
                 const unsigned short* __restrict__ Wbf,  // (DM, DM)   bf16
                 void* __restrict__ dst,
                 const float* __restrict__ bias,
                 float scale, int mode) {
  int wid  = blockIdx.x * (blockDim.x >> 5) + (threadIdx.x >> 5);
  int lane = threadIdx.x & 31;
  int ot = wid & 31;            // DM/16 = 32 o-tiles
  int lt = (wid >> 5) & 127;    // L/16 = 128 l-tiles
  int b  = wid >> 12;           // batch
  int m  = lane & 15;
  int hi = lane >> 4;

  floatx8 acc = {};
  const unsigned short* arow = Xt + ((size_t)b * L_SEQ + lt * 16 + m) * DM + hi * 8;
  const unsigned short* brow = Wbf + ((size_t)(ot * 16 + m)) * DM + hi * 8;
  #pragma unroll
  for (int k0 = 0; k0 < DM; k0 += 32) {
    BF16x16 a, w;
    a.q[0] = *(const uint4*)(arow + k0);        // K chunk lo
    a.q[1] = *(const uint4*)(arow + k0 + 16);   // K chunk hi
    w.q[0] = *(const uint4*)(brow + k0);
    w.q[1] = *(const uint4*)(brow + k0 + 16);
    acc = wmma_bf16(a, w, acc);
  }

  if (mode == 0) {              // transposed bf16: dst[b][l][o]
    unsigned short* d = (unsigned short*)dst;
    #pragma unroll
    for (int r = 0; r < 8; r++) {
      size_t row = (size_t)b * L_SEQ + lt * 16 + r + hi * 8;
      d[row * DM + ot * 16 + m] = f2bfbits(acc[r] * scale);
    }
  } else if (mode == 1) {       // normal bf16: dst[b][o][l], packed store
    union { unsigned short t[8]; uint4 q; } u;
    #pragma unroll
    for (int r = 0; r < 8; r++) u.t[r] = f2bfbits(acc[r]);
    unsigned short* d = (unsigned short*)dst +
        ((size_t)b * DM + ot * 16 + m) * L_SEQ + lt * 16 + hi * 8;
    *(uint4*)d = u.q;
  } else {                      // fp32 + bias: out[b][o][l]
    float bv = bias[ot * 16 + m];
    float* d = (float*)dst + ((size_t)b * DM + ot * 16 + m) * L_SEQ + lt * 16 + hi * 8;
    #pragma unroll
    for (int r = 0; r < 8; r++) d[r] = acc[r] + bv;
  }
}

// --------------------------- fused attention -------------------------------
// One WG per (b, h, 16-row q-tile). 4 waves. Scores strip lives in LDS.
__global__ __launch_bounds__(128)
void attn_kernel(const unsigned short* __restrict__ Qt,  // (B,L,DM) bf16, pre-scaled
                 const unsigned short* __restrict__ Kt,  // (B,L,DM) bf16
                 const unsigned short* __restrict__ Vb,  // (B,DM,L) bf16
                 float* __restrict__ Aout,               // (B,H,L,L) fp32
                 unsigned short* __restrict__ Ht) {      // (B,L,DM) bf16
  __shared__ float S[16 * SP];
  __shared__ float red[16][8];
  __shared__ float rowstat[16];

  int b = blockIdx.z, h = blockIdx.y, q0 = blockIdx.x * 16;
  int w = threadIdx.x >> 5, lane = threadIdx.x & 31;
  int m = lane & 15, hi = lane >> 4;

  // Q fragments for this q-tile (d_k = 64 -> two K=32 steps), loaded once.
  BF16x16 aQ0, aQ1;
  {
    const unsigned short* qrow =
        Qt + ((size_t)b * L_SEQ + q0 + m) * DM + h * 64 + hi * 8;
    aQ0.q[0] = *(const uint4*)(qrow);
    aQ0.q[1] = *(const uint4*)(qrow + 16);
    aQ1.q[0] = *(const uint4*)(qrow + 32);
    aQ1.q[1] = *(const uint4*)(qrow + 48);
  }

  // ---- scores: waves split the 128 column tiles ----
  for (int ct = w; ct < L_SEQ / 16; ct += 4) {
    const unsigned short* krow =
        Kt + ((size_t)b * L_SEQ + ct * 16 + m) * DM + h * 64 + hi * 8;
    BF16x16 b0, b1;
    b0.q[0] = *(const uint4*)(krow);       b0.q[1] = *(const uint4*)(krow + 16);
    b1.q[0] = *(const uint4*)(krow + 32);  b1.q[1] = *(const uint4*)(krow + 48);
    floatx8 acc = {};
    acc = wmma_bf16(aQ0, b0, acc);
    acc = wmma_bf16(aQ1, b1, acc);
    #pragma unroll
    for (int r = 0; r < 8; r++)
      S[(r + hi * 8) * SP + ct * 16 + m] = acc[r];
  }
  __syncthreads();

  // ---- softmax over each of the 16 rows (8 threads per row) ----
  int t = threadIdx.x, row = t >> 3, sub = t & 7;
  float* srow = &S[row * SP];
  float mx = -3.4e38f;
  for (int c = sub; c < L_SEQ; c += 8) mx = fmaxf(mx, srow[c]);
  red[row][sub] = mx;
  __syncthreads();
  if (sub == 0) {
    float v = red[row][0];
    #pragma unroll
    for (int i = 1; i < 8; i++) v = fmaxf(v, red[row][i]);
    rowstat[row] = v;
  }
  __syncthreads();
  float rm = rowstat[row];
  float sum = 0.f;
  for (int c = sub; c < L_SEQ; c += 8) {
    float e = __expf(srow[c] - rm);
    srow[c] = e;
    sum += e;
  }
  red[row][sub] = sum;
  __syncthreads();
  if (sub == 0) {
    float v = 0.f;
    #pragma unroll
    for (int i = 0; i < 8; i++) v += red[row][i];
    rowstat[row] = 1.0f / v;
  }
  __syncthreads();

  // ---- normalize in LDS + single coalesced write of A ----
  float* Arow = Aout + (((size_t)(b * NH + h)) * L_SEQ + q0) * L_SEQ;
  for (int idx = t; idx < 16 * L_SEQ; idx += 128) {
    int r = idx >> 11, c = idx & (L_SEQ - 1);
    float p = S[r * SP + c] * rowstat[r];
    S[r * SP + c] = p;
    Arow[(size_t)r * L_SEQ + c] = p;
  }
  __syncthreads();

  // ---- H = A @ V^T : wave w owns d-tile [w*16, w*16+16) ----
  // A-fragments built from LDS f32 with one v_perm_b32 per bf16 pair.
  floatx8 acc = {};
  const float* sErow = &S[m * SP];
  const unsigned short* vrow =
      Vb + ((size_t)b * DM + h * 64 + w * 16 + m) * L_SEQ + hi * 8;
  for (int kt = 0; kt < L_SEQ / 32; kt++) {
    BF16x16 a;
    int k1 = kt * 32 + hi * 8;
    #pragma unroll
    for (int i = 0; i < 4; i++)
      a.u[i] = pk_bf16_trunc(sErow[k1 + 2 * i], sErow[k1 + 2 * i + 1]);
    #pragma unroll
    for (int i = 0; i < 4; i++)
      a.u[4 + i] = pk_bf16_trunc(sErow[k1 + 16 + 2 * i], sErow[k1 + 16 + 2 * i + 1]);
    BF16x16 bv;
    bv.q[0] = *(const uint4*)(vrow + kt * 32);
    bv.q[1] = *(const uint4*)(vrow + kt * 32 + 16);
    acc = wmma_bf16(a, bv, acc);
  }
  #pragma unroll
  for (int r = 0; r < 8; r++) {
    Ht[((size_t)b * L_SEQ + q0 + r + hi * 8) * DM + h * 64 + w * 16 + m] =
        f2bfbits(acc[r]);
  }
}

// ------------------------------- launcher ----------------------------------
extern "C" void kernel_launch(void* const* d_in, const int* in_sizes, int n_in,
                              void* d_out, int out_size, void* d_ws, size_t ws_size,
                              hipStream_t stream) {
  (void)in_sizes; (void)n_in; (void)out_size; (void)ws_size;
  const float* Xq = (const float*)d_in[0];
  const float* Xk = (const float*)d_in[1];
  const float* Xv = (const float*)d_in[2];
  const float* Wq = (const float*)d_in[3];
  const float* Wk = (const float*)d_in[4];
  const float* Wv = (const float*)d_in[5];
  const float* Wh = (const float*)d_in[6];
  const float* bh = (const float*)d_in[7];

  float* outp = (float*)d_out;                          // (B, DM, L)
  float* Aout = outp + (size_t)NB * DM * L_SEQ;         // (B, H, L, L)

  char* ws = (char*)d_ws;
  const size_t WSZ = (size_t)DM * DM * 2;               // bf16 weight
  const size_t XSZ = (size_t)NB * L_SEQ * DM * 2;       // bf16 activation
  unsigned short* WQb = (unsigned short*)(ws + 0 * WSZ);
  unsigned short* WKb = (unsigned short*)(ws + 1 * WSZ);
  unsigned short* WVb = (unsigned short*)(ws + 2 * WSZ);
  unsigned short* WHb = (unsigned short*)(ws + 3 * WSZ);
  char* base = ws + 4 * WSZ;
  unsigned short* Xqt = (unsigned short*)(base + 0 * XSZ);
  unsigned short* Xkt = (unsigned short*)(base + 1 * XSZ);
  unsigned short* Xvt = (unsigned short*)(base + 2 * XSZ);
  unsigned short* Qt  = (unsigned short*)(base + 3 * XSZ);
  unsigned short* Kt  = (unsigned short*)(base + 4 * XSZ);
  unsigned short* Vb  = (unsigned short*)(base + 5 * XSZ);
  unsigned short* Htc = Xqt;   // reuse: Xqt dead after Q projection

  int n = DM * DM;
  wconv_kernel<<<(n + 255) / 256, 256, 0, stream>>>(Wq, WQb, n);
  wconv_kernel<<<(n + 255) / 256, 256, 0, stream>>>(Wk, WKb, n);
  wconv_kernel<<<(n + 255) / 256, 256, 0, stream>>>(Wv, WVb, n);
  wconv_kernel<<<(n + 255) / 256, 256, 0, stream>>>(Wh, WHb, n);

  dim3 tg(L_SEQ / 32, DM / 32, NB);
  xpose_kernel<<<tg, 256, 0, stream>>>(Xq, Xqt);
  xpose_kernel<<<tg, 256, 0, stream>>>(Xk, Xkt);
  xpose_kernel<<<tg, 256, 0, stream>>>(Xv, Xvt);

  int pblocks = (NB * (L_SEQ / 16) * (DM / 16)) / 8;    // 8 waves / block
  proj_kernel<<<pblocks, 256, 0, stream>>>(Xqt, WQb, Qt, nullptr, 0.125f, 0);
  proj_kernel<<<pblocks, 256, 0, stream>>>(Xkt, WKb, Kt, nullptr, 1.0f, 0);
  proj_kernel<<<pblocks, 256, 0, stream>>>(Xvt, WVb, Vb, nullptr, 1.0f, 1);

  dim3 ag(L_SEQ / 16, NH, NB);
  attn_kernel<<<ag, 128, 0, stream>>>(Qt, Kt, Vb, Aout, Htc);

  proj_kernel<<<pblocks, 256, 0, stream>>>(Htc, WHb, outp, bh, 1.0f, 2);
}
